// sLSTMcell_76536317215148
// MI455X (gfx1250) — compile-verified, matrix operand order checked by reading, not run on hardware
//
#include <hip/hip_runtime.h>
#include <hip/hip_bf16.h>
#include <math.h>

// ---------------------------------------------------------------------------
// sLSTM for MI455X (gfx1250, wave32, WMMA)
// Phase 1: Wx = x @ W^T + b as bf16 WMMA GEMM (M=16384, N=4096, K=1024),
//          32x64 per wave register blocking.
// Phase 2: persistent-kernel sequential scan; per step Rh = h @ R^T with
//          v_wmma_f32_16x16x32_bf16 reading both operands from LDS.
//          R staged once and h staged per step via GLOBAL_LOAD_ASYNC_TO_LDS
//          (ASYNCcnt), device-wide software barrier between steps.
// ---------------------------------------------------------------------------

typedef __attribute__((ext_vector_type(16))) __bf16 v16bf;
typedef __attribute__((ext_vector_type(8)))  __bf16 v8bf;
typedef __attribute__((ext_vector_type(8)))  float  v8f;

static constexpr int kB = 16;    // batch
static constexpr int kT = 1024;  // time steps
static constexpr int kD = 1024;  // hidden dim
static constexpr float kEPS = 1e-8f;

// LDS row padding: 1024 + 8 bf16 -> 2064B row stride -> bank-conflict-free
static constexpr int kLdsRow = kD + 8;
static constexpr int kRlElems = 4 * 16 * kLdsRow;            // R tiles, bf16
static constexpr int kHlElems = 16 * kLdsRow;                // h matrix, bf16
static constexpr size_t kSmemBytes =
    (size_t)(kRlElems + kHlElems) * 2 + 4 * 16 * 16 * 4;     // + rh floats

// round-to-nearest-even fp32 -> bf16
__device__ __forceinline__ unsigned short f2bf(float f) {
    unsigned u = __float_as_uint(f);
    u += 0x7FFFu + ((u >> 16) & 1u);
    return (unsigned short)(u >> 16);
}

__device__ __forceinline__ v16bf cat16(v8bf lo, v8bf hi) {
    return __builtin_shufflevector(lo, hi, 0,1,2,3,4,5,6,7,8,9,10,11,12,13,14,15);
}

// bf16 A/B fragment for 16x16x32 from a row-major buffer (global or LDS):
// per-lane K chunks at k0 + half*8 and k0 + 16 + half*8 (ISA 7.12.2 layout).
__device__ __forceinline__ v16bf load_frag(const __bf16* base, int row, int ld,
                                           int k0, int hh) {
    const v8bf lo = *(const v8bf*)(base + (size_t)row * ld + k0 + hh * 8);
    const v8bf hi = *(const v8bf*)(base + (size_t)row * ld + k0 + 16 + hh * 8);
    return cat16(lo, hi);
}

// CDNA5 async global -> LDS copy, 16B per lane, tracked by ASYNCcnt.
// vdst is the per-lane LDS byte address (low 32 bits of a generic LDS ptr).
__device__ __forceinline__ void async_copy_b128(void* lds_dst, const void* gsrc) {
    const unsigned ldsa = (unsigned)(size_t)lds_dst;
    asm volatile("global_load_async_to_lds_b128 %0, %1, off"
                 :: "v"(ldsa), "v"(gsrc) : "memory");
}
__device__ __forceinline__ void wait_asynccnt0() {
    asm volatile("s_wait_asynccnt 0x0" ::: "memory");
}

// ---------------------------------------------------------------------------
// fp32 -> bf16 conversion
// ---------------------------------------------------------------------------
__global__ void cvt_bf16_kernel(const float* __restrict__ src,
                                unsigned short* __restrict__ dst, int n) {
    int i = blockIdx.x * blockDim.x + threadIdx.x;
    if (i < n) dst[i] = f2bf(src[i]);
}

// ---------------------------------------------------------------------------
// Phase 1: Wx[t][g][b][k] = sum_d x[b][t][d] * W[g][k][d] + bias[g][k]
// One wave computes a 32(M) x 64(N) strip: 8 accumulators, B frags shared
// across both M tiles, A frags shared across the 4 N tiles.
// ---------------------------------------------------------------------------
__global__ void __launch_bounds__(256)
wx_gemm_kernel(const unsigned short* __restrict__ xb,
               const unsigned short* __restrict__ Wb,
               const float* __restrict__ bias,
               float* __restrict__ Wx) {
    const int wave  = blockIdx.x * 8 + (threadIdx.x >> 5);
    const int lane  = threadIdx.x & 31;
    const int mt2   = wave & 511;    // 512 tiles of 32 over M = B*T
    const int nt4   = wave >> 9;     // 64 strips of 64 over N = 4*D
    const int hh    = lane >> 4;
    const int l15   = lane & 15;

    const __bf16* A  = (const __bf16*)xb;   // [B*T, D] row-major
    const __bf16* Bm = (const __bf16*)Wb;   // [4*D, D] row-major (W flattened)

    v8f acc[2][4];
#pragma unroll
    for (int mi = 0; mi < 2; ++mi)
#pragma unroll
        for (int j = 0; j < 4; ++j) acc[mi][j] = (v8f)(0.0f);

    const int arow0 = mt2 * 32 + l15;
    for (int k0 = 0; k0 < kD; k0 += 32) {
        const v16bf a0 = load_frag(A, arow0,      kD, k0, hh);
        const v16bf a1 = load_frag(A, arow0 + 16, kD, k0, hh);
#pragma unroll
        for (int j = 0; j < 4; ++j) {
            const int brow = nt4 * 64 + j * 16 + l15;
            const v16bf bf = load_frag(Bm, brow, kD, k0, hh);
            acc[0][j] = __builtin_amdgcn_wmma_f32_16x16x32_bf16(
                false, a0, false, bf, (short)0, acc[0][j], false, false);
            acc[1][j] = __builtin_amdgcn_wmma_f32_16x16x32_bf16(
                false, a1, false, bf, (short)0, acc[1][j], false, false);
        }
    }

#pragma unroll
    for (int mi = 0; mi < 2; ++mi) {
#pragma unroll
        for (int j = 0; j < 4; ++j) {
            const int n    = nt4 * 64 + j * 16 + l15;
            const int g    = n >> 10;
            const int kout = n & 1023;
            const float bv = bias[g * kD + kout];
#pragma unroll
            for (int v = 0; v < 8; ++v) {
                const int m = mt2 * 32 + mi * 16 + v + hh * 8;  // m = b*T + t
                const int b = m >> 10;
                const int t = m & 1023;
                Wx[((size_t)(t * 4 + g) * kB + b) * kD + kout] = acc[mi][j][v] + bv;
            }
        }
    }
}

// ---------------------------------------------------------------------------
// Device-wide barrier (sense via generation counter)
// ---------------------------------------------------------------------------
struct GridBar { unsigned cnt; unsigned gen; };

__global__ void bar_init_kernel(GridBar* bar) { bar->cnt = 0u; bar->gen = 0u; }

__device__ __forceinline__ void grid_barrier(GridBar* bar, unsigned nb,
                                             unsigned& lgen) {
    __syncthreads();
    if (threadIdx.x == 0) {
        const unsigned old = atomicAdd(&bar->cnt, 1u);
        if (old == nb - 1u) {
            atomicExch(&bar->cnt, 0u);
            __threadfence();
            atomicAdd(&bar->gen, 1u);
        } else {
            volatile unsigned* g = &bar->gen;
            while (*g == lgen) { __builtin_amdgcn_s_sleep(4); }
        }
        __threadfence();
    }
    lgen++;
    __syncthreads();
}

// ---------------------------------------------------------------------------
// Phase 2: persistent scan. Grid = 64 blocks x 128 threads.
// Block owns columns [ko, ko+16); wave g computes gate g's 16x16 Rh tile.
// R rows and per-step h live in LDS, filled by async global->LDS copies.
// ---------------------------------------------------------------------------
__global__ void __launch_bounds__(128)
slstm_scan_kernel(const unsigned short* __restrict__ Rb,
                  const float* __restrict__ Wx,
                  unsigned short* __restrict__ hb,   // 2 x [B, D] bf16 ping-pong
                  float* __restrict__ cs_,
                  float* __restrict__ ns_,
                  float* __restrict__ ms_,
                  float* __restrict__ out,
                  GridBar* bar) {
    extern __shared__ char smem[];
    __bf16* Rl = (__bf16*)smem;                                  // [4*16][kLdsRow]
    __bf16* hl = Rl + kRlElems;                                  // [16][kLdsRow]
    float*  rh = (float*)(smem + (size_t)(kRlElems + kHlElems) * 2); // [4][16][16]

    const int tid  = threadIdx.x;
    const int gate = tid >> 5;
    const int lane = tid & 31;
    const int hh   = lane >> 4;
    const int l15  = lane & 15;
    const int ko   = blockIdx.x * 16;
    const unsigned nb = gridDim.x;

    const __bf16* Rg = (const __bf16*)Rb;   // [4*D, D] row-major

    // Stage this block's 64 R rows (4 gates x 16 output features) into LDS
    // with async global->LDS b128 copies (one row = 2KB = 128 lanes x 16B).
    for (int r = 0; r < 64; ++r) {
        const int grow = (r >> 4) * kD + ko + (r & 15);
        async_copy_b128(Rl + (size_t)r * kLdsRow + tid * 8,
                        Rg + (size_t)grow * kD + tid * 8);
    }

    // Init state + h ping buffer for the cells this block owns.
    for (int c = tid; c < 256; c += 128) {
        const int b  = c >> 4;
        const int kl = c & 15;
        const int si = b * kD + ko + kl;
        cs_[si] = 0.f; ns_[si] = 0.f; ms_[si] = 0.f;
        hb[si] = 0;  // buffer 0, bf16 zero
    }
    unsigned lgen = 0;
    wait_asynccnt0();
    __threadfence();
    grid_barrier(bar, nb, lgen);

    const __bf16* bl_row = Rl + (size_t)(gate * 16 + l15) * kLdsRow;
    const __bf16* al_row = hl + (size_t)l15 * kLdsRow;

    for (int t = 0; t < kT; ++t) {
        const float* wxt = Wx + (size_t)t * 4 * kB * kD;

        // Async-copy full h (16 x 1024 bf16 = 32KB) into LDS for all 4 waves.
        {
            const unsigned short* hp = hb + (size_t)(t & 1) * kB * kD;
            for (int r = 0; r < 16; ++r) {
                async_copy_b128(hl + (size_t)r * kLdsRow + tid * 8,
                                hp + (size_t)r * kD + tid * 8);
            }
        }
        // Warm this block's 4KB Wx slice while the copy + WMMA loop run.
        for (int c = tid; c < 256; c += 128) {
            const int b  = c >> 4;
            const int k  = ko + (c & 15);
#pragma unroll
            for (int g = 0; g < 4; ++g)
                __builtin_prefetch(&wxt[(g * kB + b) * kD + k], 0, 1);
        }
        wait_asynccnt0();
        __syncthreads();

        v8f acc = (v8f)(0.0f);
        for (int k0 = 0; k0 < kD; k0 += 32) {
            const v8bf a_lo = *(const v8bf*)(al_row + k0 + hh * 8);
            const v8bf a_hi = *(const v8bf*)(al_row + k0 + 16 + hh * 8);
            const v8bf b_lo = *(const v8bf*)(bl_row + k0 + hh * 8);
            const v8bf b_hi = *(const v8bf*)(bl_row + k0 + 16 + hh * 8);
            acc = __builtin_amdgcn_wmma_f32_16x16x32_bf16(
                false, cat16(a_lo, a_hi), false, cat16(b_lo, b_hi),
                (short)0, acc, false, false);
        }
#pragma unroll
        for (int v = 0; v < 8; ++v)
            rh[(gate * 16 + (v + hh * 8)) * 16 + l15] = acc[v];
        __syncthreads();

        // Gate math: 256 (b,k) cells per block, 2 per thread.
        for (int c = tid; c < 256; c += 128) {
            const int b  = c >> 4;
            const int kl = c & 15;
            const int k  = ko + kl;
            const int si = b * kD + k;
            const float zb  = wxt[(0 * kB + b) * kD + k] + rh[(0 * 16 + b) * 16 + kl];
            const float ib  = wxt[(1 * kB + b) * kD + k] + rh[(1 * 16 + b) * 16 + kl];
            const float fbv = wxt[(2 * kB + b) * kD + k] + rh[(2 * 16 + b) * 16 + kl];
            const float ob  = wxt[(3 * kB + b) * kD + k] + rh[(3 * 16 + b) * 16 + kl];

            const float mprev = ms_[si];
            const float cprev = cs_[si];
            const float nprev = ns_[si];

            const float mnew = fmaxf(fbv + mprev, ib);
            const float iv = expf(ib - mnew);
            const float fv = expf(fbv + mprev - mnew);
            const float z  = tanhf(zb);
            const float o  = 1.f / (1.f + expf(-ob));
            const float cnew = fv * cprev + iv * z;
            const float nnew = fv * nprev + iv;
            const float h    = o * (cnew / (nnew + kEPS));

            cs_[si] = cnew; ns_[si] = nnew; ms_[si] = mnew;
            out[(size_t)b * kT * kD + (size_t)t * kD + k] = h;   // h_seq [B,T,D]
            hb[(size_t)((t + 1) & 1) * kB * kD + si] = f2bf(h);
            if (t == kT - 1) {
                const size_t fo = (size_t)kB * kT * kD;
                out[fo + si]               = h;
                out[fo + 1 * kB * kD + si] = cnew;
                out[fo + 2 * kB * kD + si] = nnew;
                out[fo + 3 * kB * kD + si] = mnew;
            }
        }
        __threadfence();
        grid_barrier(bar, nb, lgen);
    }
}

// ---------------------------------------------------------------------------
// Launch
// ---------------------------------------------------------------------------
extern "C" void kernel_launch(void* const* d_in, const int* in_sizes, int n_in,
                              void* d_out, int out_size, void* d_ws, size_t ws_size,
                              hipStream_t stream) {
    const float* x    = (const float*)d_in[0];  // [B,T,D]
    const float* W    = (const float*)d_in[1];  // [4,D,D]
    const float* bias = (const float*)d_in[2];  // [4,D]
    const float* R    = (const float*)d_in[3];  // [4,D,D]
    float* out = (float*)d_out;

    char* ws = (char*)d_ws;
    size_t off = 0;
    float* Wx = (float*)(ws + off);              off += (size_t)kT * 4 * kB * kD * 4;
    unsigned short* xb = (unsigned short*)(ws + off); off += (size_t)kB * kT * kD * 2;
    unsigned short* Wb = (unsigned short*)(ws + off); off += (size_t)4 * kD * kD * 2;
    unsigned short* Rb = (unsigned short*)(ws + off); off += (size_t)4 * kD * kD * 2;
    unsigned short* hb = (unsigned short*)(ws + off); off += (size_t)2 * kB * kD * 2;
    float* cs = (float*)(ws + off);              off += (size_t)kB * kD * 4;
    float* ns = (float*)(ws + off);              off += (size_t)kB * kD * 4;
    float* ms = (float*)(ws + off);              off += (size_t)kB * kD * 4;
    GridBar* bar = (GridBar*)(ws + off);         off += 64;

    const int nX = kB * kT * kD;
    const int nW = 4 * kD * kD;
    cvt_bf16_kernel<<<(nX + 255) / 256, 256, 0, stream>>>(x, xb, nX);
    cvt_bf16_kernel<<<(nW + 255) / 256, 256, 0, stream>>>(W, Wb, nW);
    cvt_bf16_kernel<<<(nW + 255) / 256, 256, 0, stream>>>(R, Rb, nW);
    bar_init_kernel<<<1, 1, 0, stream>>>(bar);

    // M tiles = (B*T)/32 = 512, N strips = (4*D)/64 = 64 -> 32768 waves, 8/block
    wx_gemm_kernel<<<4096, 256, 0, stream>>>(xb, Wb, bias, Wx);

    // Persistent scan: 64 blocks (one per 16-column slice of D), 128 threads,
    // ~169KB dynamic LDS (fits in the WGP's 320KB at 1 block/WGP).
    slstm_scan_kernel<<<kD / 16, 128, kSmemBytes, stream>>>(
        Rb, Wx, hb, cs, ns, ms, out, bar);
}
